// NeuronGAT_26955214750376
// MI455X (gfx1250) — matrix operand order repaired and checked by reading, not compiled, not run on hardware
//
#include <hip/hip_runtime.h>
#include <hip/hip_fp16.h>
#include <math.h>

typedef __attribute__((ext_vector_type(16))) _Float16 v16h;
typedef __attribute__((ext_vector_type(8)))  float    v8f;
typedef __attribute__((ext_vector_type(4)))  unsigned u32x4;
typedef __attribute__((ext_vector_type(8)))  int      i32x8;
typedef __attribute__((ext_vector_type(4)))  int      i32x4;

// ---------------- helpers ----------------
__device__ __forceinline__ unsigned f2mono(float f) {
  unsigned u = __float_as_uint(f);
  return (u & 0x80000000u) ? ~u : (u | 0x80000000u);
}
__device__ __forceinline__ float mono2f(unsigned m) {
  return __uint_as_float((m & 0x80000000u) ? (m & 0x7fffffffu) : ~m);
}
__device__ __forceinline__ float gelu_exact(float v) {
  return 0.5f * v * (1.0f + erff(v * 0.7071067811865475f));
}
__device__ __forceinline__ float lrelu02(float v) { return v > 0.f ? v : 0.2f * v; }

// Build a 2D TDM descriptor and issue TENSOR_LOAD_TO_LDS (clang-23 6-arg builtin).
// D# group0: [1:0]=count=1, [63:32]=lds byte addr, [120:64]=global addr, [127:126]=type 2.
// D# group1: data_size(2B), tensor_dim0/1 (remaining extents -> OOB rows read as zero),
//            tile 32(K) x 64(rows), tensor_dim0_stride = ldK elements.
__device__ __forceinline__ void tdm_load_tile_f16(
    const _Float16* gptr, unsigned lds_off, int tdim0, int tdim1, long stride0)
{
  unsigned long long ga = (unsigned long long)(uintptr_t)gptr;
  u32x4 g0;
  g0[0] = 1u;                                      // count=1, user mode
  g0[1] = lds_off;                                 // LDS byte address
  g0[2] = (unsigned)(ga & 0xffffffffull);          // global addr [31:0]
  g0[3] = (unsigned)((ga >> 32) & 0x1ffffffull) | (2u << 30);  // addr[56:32] | type=2
  i32x8 g1;
  g1[0] = (1 << 16);                               // workgroup_mask=0, data_size=1 (2 bytes)
  g1[1] = (int)((tdim0 & 0xffff) << 16);           // tensor_dim0[15:0]
  g1[2] = (int)(((unsigned)tdim0 >> 16) & 0xffffu) | ((tdim1 & 0xffff) << 16);
  g1[3] = (int)((((unsigned)tdim1 >> 16) & 0xffffu) | (32u << 16));  // tile_dim0 = 32
  g1[4] = 64;                                      // tile_dim1 = 64, tile_dim2 = 0
  g1[5] = (int)(stride0 & 0xffffffffll);           // tensor_dim0_stride lo
  g1[6] = (int)((stride0 >> 32) & 0xffffll);       // stride hi | dim1_stride lo = 0
  g1[7] = 0;
  i32x4 gz4 = {0, 0, 0, 0};                        // 2D: groups 2/3 unused
  i32x8 gz8 = {0, 0, 0, 0, 0, 0, 0, 0};
  __builtin_amdgcn_tensor_load_to_lds(g0, g1, gz4, gz4, gz8, 0);
}

// ---------------- fills ----------------
__global__ void k_fill_f32(float* p, float v, int n) {
  int i = blockIdx.x * blockDim.x + threadIdx.x;
  if (i < n) p[i] = v;
}
__global__ void k_fill_u32(unsigned* p, unsigned v, int n) {
  int i = blockIdx.x * blockDim.x + threadIdx.x;
  if (i < n) p[i] = v;
}

// ---------------- f32 -> f16 converts (feed the TDM GEMM) ----------------
__global__ void k_cvt_f16(const float* __restrict__ x, _Float16* __restrict__ y, int n) {
  int i = blockIdx.x * blockDim.x + threadIdx.x;
  if (i < n) y[i] = (_Float16)x[i];
}
__global__ void k_cvt_f16_T(const float* __restrict__ B, _Float16* __restrict__ BT,
                            int K, int N) {
  int i = blockIdx.x * blockDim.x + threadIdx.x;
  if (i >= K * N) return;
  int k = i / N, n = i % N;
  BT[(size_t)n * K + k] = (_Float16)B[i];
}

// ---------------- TDM + WMMA GEMM ----------------
// C[M,N](f32,ldc) = A16[M,K] @ BT16[N,K]^T (+bias).
// 256 threads = 8 waves; 64x64 tile; wave -> 16x32 strip (2 wmma acc); K step 32.
// Tiles staged by the Tensor Data Mover, double-buffered on TENSORcnt.
__global__ __launch_bounds__(256) void k_gemm_wmma_tdm(
    const _Float16* __restrict__ A16, const _Float16* __restrict__ BT16,
    const float* __restrict__ bias, float* __restrict__ C,
    int M, int N, int K, int ldc)
{
  __shared__ _Float16 As[2][64 * 32];
  __shared__ _Float16 Bs[2][64 * 32];
  const int tid  = threadIdx.x;
  const int wave = tid >> 5;
  const int lane = tid & 31;
  const int tm = blockIdx.y * 64;
  const int tn = blockIdx.x * 64;
  const int wr = wave & 3;          // 16-row strip
  const int wc = wave >> 2;         // 32-col group
  const int m16 = lane & 15;
  const int hi  = lane >> 4;
  const int nk = K >> 5;            // K is always a multiple of 32 here

  const _Float16* aTile = &A16[(size_t)tm * K];
  const _Float16* bTile = &BT16[(size_t)tn * K];

  if (wave == 0) {                  // one wave drives the TDM
    tdm_load_tile_f16(aTile, (unsigned)(uintptr_t)&As[0][0], K, M - tm, K);
    tdm_load_tile_f16(bTile, (unsigned)(uintptr_t)&Bs[0][0], K, N - tn, K);
  }

  v8f acc0 = {};
  v8f acc1 = {};

  for (int kt = 0; kt < nk; ++kt) {
    const int buf = kt & 1;
    if (wave == 0) {
      if (kt + 1 < nk) {
        const int k0n = (kt + 1) << 5;
        tdm_load_tile_f16(aTile + k0n, (unsigned)(uintptr_t)&As[buf ^ 1][0],
                          K - k0n, M - tm, K);
        tdm_load_tile_f16(bTile + k0n, (unsigned)(uintptr_t)&Bs[buf ^ 1][0],
                          K - k0n, N - tn, K);
        __builtin_amdgcn_s_wait_tensorcnt(2);  // in-order: current buffer complete
      } else {
        __builtin_amdgcn_s_wait_tensorcnt(0);
      }
    }
    __syncthreads();

    // A fragment: lane = M row; VGPR0-3 hold K pairs kb..kb+7, VGPR4-7 hold 8+kb.. (kb = 16*hi)
    v16h af, bf0, bf1;
    const _Float16* ap = &As[buf][(wr * 16 + m16) * 32];
    const int kb = hi * 16;
    #pragma unroll
    for (int p = 0; p < 4; ++p) {
      af[2*p]     = ap[kb + 2*p];
      af[2*p + 1] = ap[kb + 2*p + 1];
      af[8 + 2*p]     = ap[8 + kb + 2*p];
      af[8 + 2*p + 1] = ap[8 + kb + 2*p + 1];
    }
    // B fragment: lane = N col; halves = 16 consecutive K (lower/upper half-wave)
    const _Float16* bp0 = &Bs[buf][(wc * 32 + m16) * 32 + hi * 16];
    const _Float16* bp1 = &Bs[buf][(wc * 32 + 16 + m16) * 32 + hi * 16];
    #pragma unroll
    for (int j = 0; j < 16; ++j) { bf0[j] = bp0[j]; bf1[j] = bp1[j]; }

    acc0 = __builtin_amdgcn_wmma_f32_16x16x32_f16(false, af, false, bf0,
                                                  (short)0, acc0, false, false);
    acc1 = __builtin_amdgcn_wmma_f32_16x16x32_f16(false, af, false, bf1,
                                                  (short)0, acc1, false, false);
    __syncthreads();   // all waves done with buf before wave0 overwrites it next iter
  }

  // C/D layout: VGPR v -> M = v + 8*hi ; N = lane&15
  const int rowb = tm + wr * 16 + hi * 8;
  const int col0 = tn + wc * 32 + m16;
  const int col1 = col0 + 16;
  #pragma unroll
  for (int v = 0; v < 8; ++v) {
    int gm = rowb + v;
    if (gm < M) {
      if (col0 < N) C[(size_t)gm * ldc + col0] = acc0[v] + (bias ? bias[col0] : 0.f);
      if (col1 < N) C[(size_t)gm * ldc + col1] = acc1[v] + (bias ? bias[col1] : 0.f);
    }
  }
}

// ---------------- LayerNorm (in-place safe), one block per row ----------------
__global__ __launch_bounds__(256) void k_layernorm(
    const float* __restrict__ x, const float* __restrict__ g,
    const float* __restrict__ b, float* __restrict__ y, int D)
{
  __shared__ float s1[256], s2[256];
  int row = blockIdx.x, tid = threadIdx.x;
  float a = 0.f, q = 0.f;
  for (int j = tid; j < D; j += 256) { float v = x[(size_t)row * D + j]; a += v; q += v * v; }
  s1[tid] = a; s2[tid] = q; __syncthreads();
  for (int s = 128; s; s >>= 1) {
    if (tid < s) { s1[tid] += s1[tid + s]; s2[tid] += s2[tid + s]; }
    __syncthreads();
  }
  float mean = s1[0] / D;
  float var  = s2[0] / D - mean * mean;
  float inv  = rsqrtf(var + 1e-5f);
  for (int j = tid; j < D; j += 256) {
    float v = x[(size_t)row * D + j];
    y[(size_t)row * D + j] = (v - mean) * inv * g[j] + b[j];
  }
}

// ---------------- GAT attention scores es/ed: one block per node, C=128 ----------------
__global__ __launch_bounds__(128) void k_att_scores(
    const float* __restrict__ feat, const float* __restrict__ as,
    const float* __restrict__ ad, float* __restrict__ es, float* __restrict__ ed, int H)
{
  __shared__ float r1[128], r2[128];
  int node = blockIdx.x, tid = threadIdx.x;
  for (int h = 0; h < H; ++h) {
    float v = feat[(size_t)node * H * 128 + h * 128 + tid];
    r1[tid] = v * as[h * 128 + tid];
    r2[tid] = v * ad[h * 128 + tid];
    __syncthreads();
    for (int s = 64; s; s >>= 1) {
      if (tid < s) { r1[tid] += r1[tid + s]; r2[tid] += r2[tid + s]; }
      __syncthreads();
    }
    if (tid == 0) { es[node * H + h] = r1[0]; ed[node * H + h] = r2[0]; }
    __syncthreads();
  }
}

// ---------------- edge pass 1: segment max (monotone-uint atomicMax) ----------------
__global__ void k_edge_max(const int* __restrict__ ei, const float* __restrict__ es,
                           const float* __restrict__ ed, unsigned* __restrict__ m,
                           int E, int Etot, int H)
{
  int idx = blockIdx.x * blockDim.x + threadIdx.x;
  int e = idx / H, h = idx % H;
  if (e >= Etot) return;
  int s, d;
  if (e < E) { s = ei[e]; d = ei[E + e]; } else { s = d = e - E; }
  float v = lrelu02(es[s * H + h] + ed[d * H + h]);
  atomicMax(&m[d * H + h], f2mono(v));
}

// ---------------- edge pass 2: exp & segment sum ----------------
__global__ void k_edge_expsum(const int* __restrict__ ei, const float* __restrict__ es,
                              const float* __restrict__ ed, const unsigned* __restrict__ m,
                              float* __restrict__ ssum, int E, int Etot, int H)
{
  int idx = blockIdx.x * blockDim.x + threadIdx.x;
  int e = idx / H, h = idx % H;
  if (e >= Etot) return;
  int s, d;
  if (e < E) { s = ei[e]; d = ei[E + e]; } else { s = d = e - E; }
  float v = lrelu02(es[s * H + h] + ed[d * H + h]);
  atomicAdd(&ssum[d * H + h], __expf(v - mono2f(m[d * H + h])));
}

// ---------------- edge pass 3: weighted scatter (4 features per thread) ----------------
__global__ void k_edge_scatter(const int* __restrict__ ei, const float* __restrict__ es,
                               const float* __restrict__ ed, const unsigned* __restrict__ m,
                               const float* __restrict__ ssum, const float* __restrict__ feat,
                               float* __restrict__ agg, int E, int Etot, int H)
{
  const int per = H * 32;                 // chunks of 4 over H*128 features
  int idx = blockIdx.x * blockDim.x + threadIdx.x;
  int e = idx / per;
  if (e >= Etot) return;
  int rem = idx % per;
  int h  = rem / 32;
  int c0 = (rem % 32) * 4;
  int s, d;
  if (e < E) { s = ei[e]; d = ei[E + e]; } else { s = d = e - E; }
  float v = lrelu02(es[s * H + h] + ed[d * H + h]);
  float a = __expf(v - mono2f(m[d * H + h])) / (ssum[d * H + h] + 1e-16f);
  size_t bs = (size_t)s * H * 128 + h * 128 + c0;
  size_t bd = (size_t)d * H * 128 + h * 128 + c0;
  #pragma unroll
  for (int t = 0; t < 4; ++t) atomicAdd(&agg[bd + t], a * feat[bs + t]);
}

// ---------------- elementwise ----------------
__global__ void k_bias_bn(float* __restrict__ x, const float* __restrict__ gb,
                          const float* __restrict__ bg, const float* __restrict__ bb,
                          int n, int D)
{
  int i = blockIdx.x * blockDim.x + threadIdx.x;
  if (i >= n) return;
  int c = i % D;
  x[i] = (x[i] + gb[c]) * (bg[c] * rsqrtf(1.0f + 1e-5f)) + bb[c];
}
__global__ void k_add_gelu(const float* __restrict__ h, const float* __restrict__ r,
                           float* __restrict__ y, int n)
{
  int i = blockIdx.x * blockDim.x + threadIdx.x;
  if (i >= n) return;
  float v = h[i] + (r ? r[i] : 0.f);
  y[i] = gelu_exact(v);
}
__global__ void k_gate_mul(float* __restrict__ x, const float* __restrict__ g, int n)
{
  int i = blockIdx.x * blockDim.x + threadIdx.x;
  if (i >= n) return;
  x[i] = x[i] / (1.0f + __expf(-g[i]));   // x * sigmoid(g)
}
__global__ void k_copy_cols(const float* __restrict__ src, float* __restrict__ dst,
                            int n, int D, int ldd)
{
  int i = blockIdx.x * blockDim.x + threadIdx.x;
  if (i >= n) return;
  int r = i / D, c = i % D;
  dst[(size_t)r * ldd + c] = src[i];
}
__global__ void k_logsoftmax(const float* __restrict__ x, float* __restrict__ y, int M, int D)
{
  int r = blockIdx.x * blockDim.x + threadIdx.x;
  if (r >= M) return;
  float mx = -1e30f;
  for (int d = 0; d < D; ++d) mx = fmaxf(mx, x[(size_t)r * D + d]);
  float s = 0.f;
  for (int d = 0; d < D; ++d) s += __expf(x[(size_t)r * D + d] - mx);
  float ls = __logf(s);
  for (int d = 0; d < D; ++d) y[(size_t)r * D + d] = x[(size_t)r * D + d] - mx - ls;
}

// ---------------- MHSA: one wave per (head, query); hd=32 == wave32 ----------------
__global__ __launch_bounds__(256) void k_attn(
    const float* __restrict__ Q, const float* __restrict__ Km,
    const float* __restrict__ V, float* __restrict__ Out, int Nn)
{
  const int HD = 32, DM = 128;
  int h = blockIdx.y;
  int wave = threadIdx.x >> 5, lane = threadIdx.x & 31;
  int qi = blockIdx.x * 8 + wave;
  __shared__ float qs[8][32];
  __shared__ float accs[8][32][33];
  qs[wave][lane] = Q[(size_t)qi * DM + h * HD + lane] * 0.17677669529663689f; // 32^-0.5
  __syncthreads();

  float mcur = -1e30f, l = 0.f;
  float acc[32];
  #pragma unroll
  for (int d = 0; d < 32; ++d) acc[d] = 0.f;

  for (int m0 = lane; m0 < Nn; m0 += 32) {     // lanes stripe the keys
    const float* kr = &Km[(size_t)m0 * DM + h * HD];
    const float* vr = &V[(size_t)m0 * DM + h * HD];
    __builtin_prefetch(kr + 32 * DM, 0, 0);    // speculative next-key prefetch
    __builtin_prefetch(vr + 32 * DM, 0, 0);
    float s = 0.f;
    #pragma unroll
    for (int d = 0; d < 32; ++d) s += qs[wave][d] * kr[d];
    float mn = fmaxf(mcur, s);
    float corr = __expf(mcur - mn);
    float p = __expf(s - mn);
    l = l * corr + p;
    #pragma unroll
    for (int d = 0; d < 32; ++d) acc[d] = acc[d] * corr + p * vr[d];
    mcur = mn;
  }
  float Mx = mcur;
  #pragma unroll
  for (int off = 16; off; off >>= 1) Mx = fmaxf(Mx, __shfl_xor(Mx, off, 32));
  float sc = __expf(mcur - Mx);
  float L = l * sc;
  #pragma unroll
  for (int off = 16; off; off >>= 1) L += __shfl_xor(L, off, 32);
  #pragma unroll
  for (int d = 0; d < 32; ++d) accs[wave][lane][d] = acc[d] * sc;
  __syncthreads();
  float o = 0.f;
  #pragma unroll
  for (int j = 0; j < 32; ++j) o += accs[wave][j][lane];
  Out[(size_t)qi * DM + h * HD + lane] = o / L;
}

// ---------------- orchestration ----------------
extern "C" void kernel_launch(void* const* d_in, const int* in_sizes, int n_in,
                              void* d_out, int out_size, void* d_ws, size_t ws_size,
                              hipStream_t stream) {
  (void)n_in; (void)out_size; (void)ws_size;
  const int NN   = in_sizes[0] / 256;     // 4096 nodes
  const int E    = in_sizes[1] / 2;       // 131072 edges
  const int Etot = E + NN;                // + self loops

  const float* x  = (const float*)d_in[0];
  const int*   ei = (const int*)d_in[1];
  int pi = 2;
  auto F = [&](void) { return (const float*)d_in[pi++]; };
  const float* in_ln_g = F(); const float* in_ln_b = F();
  struct Layer { const float *W,*as,*ad,*b,*bng,*bnb,*lng,*lnb,*resW; } L[3];
  for (int i = 0; i < 3; ++i) {
    L[i].W = F(); L[i].as = F(); L[i].ad = F(); L[i].b = F();
    L[i].bng = F(); L[i].bnb = F(); L[i].lng = F(); L[i].lnb = F(); L[i].resW = F();
  }
  const float* gate_W = F(); const float* gate_b = F();
  const float* dim0_W = F(); const float* dim1_W = F();
  const float* jk_W   = F(); const float* jk_b   = F();
  const float* qW = F(); const float* qb_ = F();
  const float* kW = F(); const float* kb_ = F();
  const float* vW = F(); const float* vb_ = F();
  const float* oW = F(); const float* ob_ = F();
  const float* mlp0W = F(); const float* mlp0b = F();
  const float* mln0g = F(); const float* mln0b = F();
  const float* mlp1W = F(); const float* mlp1b = F();
  const float* mln1g = F(); const float* mln1b = F();
  const float* mlp2W = F(); const float* mlp2b = F();

  // workspace carve (floats)
  float* base = (float*)d_ws;
  size_t off = 0;
  auto take = [&](size_t n) { float* p = base + off; off += n; return p; };
  float* xln   = take((size_t)NN * 256);
  float* O0    = take((size_t)NN * 512);
  float* O1    = take((size_t)NN * 512);
  float* O2    = take((size_t)NN * 128);
  float* feat  = take((size_t)NN * 512);
  float* agg   = take((size_t)NN * 512);
  float* resb  = take((size_t)NN * 512);
  float* esb   = take((size_t)NN * 4);
  float* edb   = take((size_t)NN * 4);
  unsigned* mb = (unsigned*)take((size_t)NN * 4);
  float* sb    = take((size_t)NN * 4);
  float* jkin  = take((size_t)NN * 384);
  float* xh    = take((size_t)NN * 128);
  float* qbuf  = take((size_t)NN * 128);
  float* kbuf  = take((size_t)NN * 128);
  float* vbuf  = take((size_t)NN * 128);
  float* abuf  = take((size_t)NN * 128);
  float* xh2   = take((size_t)NN * 128);
  float* mh    = take((size_t)NN * 256);
  float* mh3   = take((size_t)NN * 128);
  float* logit = take((size_t)NN * 16);
  _Float16* a16  = (_Float16*)take((size_t)NN * 256);   // M*K halves (<= 4096*512)
  _Float16* b16t = (_Float16*)take((size_t)512 * 512);  // K*N halves (<= 512*512)

  auto blocks = [](int n) { return (n + 255) / 256; };

  // gemm16: f32 inputs -> f16 (B transposed) -> TDM/WMMA GEMM
  auto gemm16 = [&](const float* A, const float* B, const float* bias, float* C,
                    int M, int N, int K, int ldc) {
    k_cvt_f16  <<<blocks(M * K), 256, 0, stream>>>(A, a16, M * K);
    k_cvt_f16_T<<<blocks(K * N), 256, 0, stream>>>(B, b16t, K, N);
    dim3 g((N + 63) / 64, (M + 63) / 64);
    k_gemm_wmma_tdm<<<g, 256, 0, stream>>>(a16, b16t, bias, C, M, N, K, ldc);
  };

  // 1. input LayerNorm
  k_layernorm<<<NN, 256, 0, stream>>>(x, in_ln_g, in_ln_b, xln, 256);

  const float* x_in = xln;
  int din = 256;
  float* outs[3] = {O0, O1, O2};
  const int Hs[3] = {4, 4, 1};

  for (int i = 0; i < 3; ++i) {
    const int H = Hs[i], od = H * 128;
    gemm16(x_in, L[i].resW, nullptr, resb, NN, od, din, od);
    gemm16(x_in, L[i].W,    nullptr, feat, NN, od, din, od);
    k_att_scores<<<NN, 128, 0, stream>>>(feat, L[i].as, L[i].ad, esb, edb, H);
    k_fill_u32<<<blocks(NN * H), 256, 0, stream>>>(mb, 0x007fffffu /* mono(-inf) */, NN * H);
    k_fill_f32<<<blocks(NN * H), 256, 0, stream>>>(sb, 0.f, NN * H);
    k_fill_f32<<<blocks(NN * od), 256, 0, stream>>>(agg, 0.f, NN * od);
    k_edge_max   <<<blocks(Etot * H), 256, 0, stream>>>(ei, esb, edb, mb, E, Etot, H);
    k_edge_expsum<<<blocks(Etot * H), 256, 0, stream>>>(ei, esb, edb, mb, sb, E, Etot, H);
    k_edge_scatter<<<blocks(Etot * H * 32), 256, 0, stream>>>(ei, esb, edb, mb, sb, feat, agg,
                                                              E, Etot, H);
    k_bias_bn<<<blocks(NN * od), 256, 0, stream>>>(agg, L[i].b, L[i].bng, L[i].bnb, NN * od, od);
    k_layernorm<<<NN, 256, 0, stream>>>(agg, L[i].lng, L[i].lnb, agg, od);
    k_add_gelu<<<blocks(NN * od), 256, 0, stream>>>(agg, resb, outs[i], NN * od);
    if (i < 2) {   // feature gate
      gemm16(outs[i], gate_W, gate_b, feat, NN, 512, 512, 512);
      k_gate_mul<<<blocks(NN * 512), 256, 0, stream>>>(outs[i], feat, NN * 512);
    }
    x_in = outs[i];
    din = od;
  }

  // JumpingKnowledge cat + projection
  gemm16(O0, dim0_W, nullptr, jkin,       NN, 128, 512, 384);
  gemm16(O1, dim1_W, nullptr, jkin + 128, NN, 128, 512, 384);
  k_copy_cols<<<blocks(NN * 128), 256, 0, stream>>>(O2, jkin + 256, NN * 128, 128, 384);
  gemm16(jkin, jk_W, jk_b, xh, NN, 128, 384, 128);

  // MHSA
  gemm16(xh, qW, qb_, qbuf, NN, 128, 128, 128);
  gemm16(xh, kW, kb_, kbuf, NN, 128, 128, 128);
  gemm16(xh, vW, vb_, vbuf, NN, 128, 128, 128);
  k_attn<<<dim3(NN / 8, 4), 256, 0, stream>>>(qbuf, kbuf, vbuf, abuf, NN);
  gemm16(abuf, oW, ob_, xh2, NN, 128, 128, 128);

  // classification MLP
  gemm16(xh2, mlp0W, mlp0b, mh, NN, 256, 128, 256);
  k_layernorm<<<NN, 256, 0, stream>>>(mh, mln0g, mln0b, mh, 256);
  k_add_gelu<<<blocks(NN * 256), 256, 0, stream>>>(mh, nullptr, mh, NN * 256);
  gemm16(mh, mlp1W, mlp1b, mh3, NN, 128, 256, 128);
  k_layernorm<<<NN, 256, 0, stream>>>(mh3, mln1g, mln1b, mh3, 128);
  k_add_gelu<<<blocks(NN * 128), 256, 0, stream>>>(mh3, nullptr, mh3, NN * 128);
  gemm16(mh3, mlp2W, mlp2b, logit, NN, 10, 128, 10);
  k_logsoftmax<<<blocks(NN), 256, 0, stream>>>(logit, (float*)d_out, NN, 10);
}
